// SkeletonGCN_34608846471746
// MI455X (gfx1250) — matrix operand order repaired and checked by reading, not compiled, not run on hardware
//
#include <hip/hip_runtime.h>

// ---------------------------------------------------------------------------
// Fused Skeleton-GCN for gfx1250 (MI455X), wave32 + WMMA bf16.
//   x = gelu(skel @ w_in + b_in)                    (B,17,128)
//   x = gelu(LN(ADJ@x @ W1 + b1))                   (B,17,256)
//   x =      LN(ADJ@x @ W2 + b2)                    (B,17,256)
// One workgroup = 8 items (144 flattened rows = 9 exact M-tiles); all
// intermediates live in two 72KB bf16 LDS ping-pong buffers (~147KB total,
// 2 WGs / 320KB WGP). Channel GEMMs use v_wmma_f32_16x16x32_bf16 with the
// weight (B) fragment loaded once per (nt,kk) and reused across all 9 M-tiles.
// The 17x17 normalized adjacency is applied in SPARSE form (49 nnz, max
// degree 5) with 2-wide bf16 channel vectorization.
// ---------------------------------------------------------------------------

#define B_TOTAL 32768
#define J       17
#define MID     128
#define CH      256
#define ITEMS   8
#define RSTRIDE 18           // rows per item in the flattened buffers (17 + 1 pad)
#define RPAD    144          // padded row count: 9 M-tiles of 16 (= ITEMS*RSTRIDE)
#define MT      (RPAD / 16)  // 9
#define NT      (CH / 16)    // 16
#define MAXDEG  8            // padded neighbor-list length (true max is 5)
#define NTHREADS 256
#define EPS     1e-5f

typedef __attribute__((ext_vector_type(2)))  __bf16 v2bf;
typedef __attribute__((ext_vector_type(4)))  __bf16 v4bf;
typedef __attribute__((ext_vector_type(8)))  __bf16 v8bf;
typedef __attribute__((ext_vector_type(16))) __bf16 v16bf;
typedef __attribute__((ext_vector_type(8)))  float  v8f;

__device__ __forceinline__ float gelu_exact(float x) {
    return 0.5f * x * (1.0f + erff(x * 0.70710678118654752f));
}

// ---- weight fp32 -> bf16 staging (runs once per launch, deterministic) ----
__global__ __launch_bounds__(256) void cvt_weights(const float* __restrict__ w1,
                                                   const float* __restrict__ w2,
                                                   __bf16* __restrict__ ws) {
    int i = blockIdx.x * 256 + threadIdx.x;
    if (i < MID * CH) ws[i] = (__bf16)w1[i];
    if (i < CH * CH)  ws[MID * CH + i] = (__bf16)w2[i];
}

// ---- WMMA tile GEMM: D(bf16 LDS) = A(bf16 LDS) @ Bw(bf16 global) + bias ----
// nt outer (striped over waves), kk middle, mt inner: each 32x16 B fragment is
// loaded once and reused by 9 WMMAs; 9 v8f accumulators live in registers.
static __device__ __forceinline__ void gemm_tiles(const __bf16* __restrict__ A,
                                                  const __bf16* __restrict__ Bw,
                                                  const float*  __restrict__ bias,
                                                  __bf16* __restrict__ D,
                                                  int kiters, int tid) {
    const int wave  = tid >> 5;
    const int lane  = tid & 31;
    const int lhalf = lane >> 4;     // 0: lanes 0-15, 1: lanes 16-31
    const int lmod  = lane & 15;
    for (int nt = wave; nt < NT; nt += 8) {
        v8f acc[MT];
        #pragma unroll
        for (int mt = 0; mt < MT; ++mt) acc[mt] = (v8f){};
        for (int kk = 0; kk < kiters; ++kk) {
            // B fragment: 32x16 bf16 tile from global weights (N-stride = CH); lane column lmod, element s -> k = kk*32 + lhalf*16 + s
            v16bf b;
            #pragma unroll
            for (int s = 0; s < 16; ++s)
                b[s] = Bw[(kk * 32 + lhalf * 16 + s) * CH + nt * 16 + lmod];
            #pragma unroll
            for (int mt = 0; mt < MT; ++mt) {
                // A fragment: 16x32 bf16 tile, rows mt*16.., K block kk*32..
                const v16bf a = *(const v16bf*)(A + (mt * 16 + lmod) * CH + kk * 32 + lhalf * 16);
                acc[mt] = __builtin_amdgcn_wmma_f32_16x16x32_bf16(
                    false, a, false, b, (short)0, acc[mt], false, false);
            }
        }
        // D layout: VGPR i -> row mt*16 + i + 8*lhalf, column nt*16 + lmod
        const float bv = bias[nt * 16 + lmod];
        #pragma unroll
        for (int mt = 0; mt < MT; ++mt) {
            const int rb = mt * 16 + (lhalf << 3);
            #pragma unroll
            for (int i = 0; i < 8; ++i)
                D[(rb + i) * CH + nt * 16 + lmod] = (__bf16)(acc[mt][i] + bv);
        }
    }
}

// ---- sparse 17x17 normalized-adjacency mix, 2 channels per iteration ----
static __device__ __forceinline__ void adj_mix_sparse(const int* __restrict__ ncnt,
                                                      const int   (*__restrict__ nidx)[MAXDEG],
                                                      const float (*__restrict__ nwgt)[MAXDEG],
                                                      const __bf16* __restrict__ S,
                                                      __bf16* __restrict__ D,
                                                      int ncol, int tid) {
    const int half = ncol >> 1;
    for (int idx = tid; idx < ITEMS * J * half; idx += NTHREADS) {
        const int item = idx / (J * half);
        const int r    = idx % (J * half);
        const int n    = r / half;
        const int c    = (r % half) * 2;
        const int base = item * RSTRIDE;
        float s0 = 0.0f, s1 = 0.0f;
        const int cnt = ncnt[n];
        for (int k = 0; k < cnt; ++k) {
            const int   m = nidx[n][k];
            const float w = nwgt[n][k];
            const v2bf  p = *(const v2bf*)(S + (base + m) * CH + c);
            s0 += w * (float)p.x;
            s1 += w * (float)p.y;
        }
        v2bf o;
        o.x = (__bf16)s0;
        o.y = (__bf16)s1;
        *(v2bf*)(D + (base + n) * CH + c) = o;
    }
}

// ---- per-item LayerNorm statistics: one wave per item, b128 LDS loads ----
static __device__ __forceinline__ void ln_stats(const __bf16* __restrict__ Z,
                                                float* __restrict__ muA,
                                                float* __restrict__ rsA,
                                                int tid) {
    const int wave = tid >> 5;   // 8 waves <-> 8 items
    const int lane = tid & 31;
    float s = 0.0f, q = 0.0f;
    for (int i = lane; i < J * (CH / 8); i += 32) {   // 544 vectors of 8 bf16
        const int row = i >> 5;                        // CH/8 == 32 per row
        const int c8  = (i & 31) * 8;
        const v8bf v = *(const v8bf*)(Z + (wave * RSTRIDE + row) * CH + c8);
        #pragma unroll
        for (int j = 0; j < 8; ++j) {
            const float f = (float)v[j];
            s += f;
            q += f * f;
        }
    }
    #pragma unroll
    for (int off = 16; off > 0; off >>= 1) {
        s += __shfl_xor(s, off, 32);
        q += __shfl_xor(q, off, 32);
    }
    if (lane == 0) {
        const float inv = 1.0f / (float)(J * CH);
        const float mu  = s * inv;
        muA[wave] = mu;
        rsA[wave] = rsqrtf(q * inv - mu * mu + EPS);
    }
}

__global__ __launch_bounds__(NTHREADS) void gcn_fused(
    const float* __restrict__ skel,
    const float* __restrict__ w_in, const float* __restrict__ b_in,
    const __bf16* __restrict__ W1b, const float* __restrict__ b1,
    const float* __restrict__ g1,   const float* __restrict__ be1,
    const __bf16* __restrict__ W2b, const float* __restrict__ b2,
    const float* __restrict__ g2,   const float* __restrict__ be2,
    float* __restrict__ out) {

    __shared__ __align__(32) __bf16 Pb[RPAD * CH];   // 72 KB
    __shared__ __align__(32) __bf16 Qb[RPAD * CH];   // 72 KB
    __shared__ float adjm[J * J];
    __shared__ float dtmp[J];
    __shared__ int   ncnt[J];
    __shared__ int   nidx[J][MAXDEG];
    __shared__ float nwgt[J][MAXDEG];
    __shared__ float muA[ITEMS], rsA[ITEMS];

    const int tid   = threadIdx.x;
    const int item0 = blockIdx.x * ITEMS;

    // ---- build normalized adjacency + sparse lists in LDS (once per WG) ----
    if (tid < J * J) {
        const int i = tid / J, j = tid % J;
        const int e0[16] = {0,1,2,0,4,5,0,7,8,9, 8,11,12, 8,14,15};
        const int e1[16] = {1,2,3,4,5,6,7,8,9,10,11,12,13,14,15,16};
        float a = (i == j) ? 1.0f : 0.0f;
        #pragma unroll
        for (int e = 0; e < 16; ++e)
            if ((i == e0[e] && j == e1[e]) || (i == e1[e] && j == e0[e])) a = 1.0f;
        adjm[tid] = a;
    }
    __syncthreads();
    if (tid < J) {
        float s = 0.0f;
        #pragma unroll
        for (int j = 0; j < J; ++j) s += adjm[tid * J + j];
        dtmp[tid] = rsqrtf(s);
    }
    __syncthreads();
    if (tid < J * J) adjm[tid] *= dtmp[tid / J] * dtmp[tid % J];
    __syncthreads();
    if (tid < J) {                      // compress row tid to (index, weight) list
        int cnt = 0;
        for (int m = 0; m < J; ++m) {
            const float w = adjm[tid * J + m];
            if (w != 0.0f) {
                nidx[tid][cnt] = m;
                nwgt[tid][cnt] = w;
                ++cnt;
            }
        }
        ncnt[tid] = cnt;
    }
    __syncthreads();

    // ---- stage 0: x0 = gelu(skel @ w_in + b_in) -> P (cols 0..127) ----
    for (int idx = tid; idx < ITEMS * J * MID; idx += NTHREADS) {
        const int item  = idx / (J * MID);
        const int r     = idx % (J * MID);
        const int joint = r / MID;
        const int c     = r % MID;
        const float* sp = skel + ((size_t)(item0 + item) * J + joint) * 3;
        float v = b_in[c] + sp[0] * w_in[0 * MID + c]
                          + sp[1] * w_in[1 * MID + c]
                          + sp[2] * w_in[2 * MID + c];
        Pb[(item * RSTRIDE + joint) * CH + c] = (__bf16)gelu_exact(v);
    }
    __syncthreads();

    // ---- layer 1 ----
    adj_mix_sparse(ncnt, nidx, nwgt, Pb, Qb, MID, tid);  // y1 = ADJ @ x0   -> Q
    __syncthreads();
    gemm_tiles(Qb, W1b, b1, Pb, MID / 32, tid);          // z1 = y1@W1 + b1 -> P (WMMA)
    __syncthreads();
    ln_stats(Pb, muA, rsA, tid);
    __syncthreads();
    // a1 = gelu(LN(z1)) -> Q, 4 channels per iteration
    for (int idx = tid; idx < ITEMS * J * (CH / 4); idx += NTHREADS) {
        const int item  = idx / (J * (CH / 4));
        const int r     = idx % (J * (CH / 4));
        const int joint = r / (CH / 4);
        const int c     = (r % (CH / 4)) * 4;
        const float mu = muA[item], rs = rsA[item];
        const v4bf   z = *(const v4bf*)&Pb[(item * RSTRIDE + joint) * CH + c];
        const float4 g = *(const float4*)&g1[joint * CH + c];
        const float4 b = *(const float4*)&be1[joint * CH + c];
        v4bf o;
        o[0] = (__bf16)gelu_exact(((float)z[0] - mu) * rs * g.x + b.x);
        o[1] = (__bf16)gelu_exact(((float)z[1] - mu) * rs * g.y + b.y);
        o[2] = (__bf16)gelu_exact(((float)z[2] - mu) * rs * g.z + b.z);
        o[3] = (__bf16)gelu_exact(((float)z[3] - mu) * rs * g.w + b.w);
        *(v4bf*)&Qb[(item * RSTRIDE + joint) * CH + c] = o;
    }
    __syncthreads();

    // ---- layer 2 ----
    adj_mix_sparse(ncnt, nidx, nwgt, Qb, Pb, CH, tid);   // y2 = ADJ @ a1   -> P
    __syncthreads();
    gemm_tiles(Pb, W2b, b2, Qb, CH / 32, tid);           // z2 = y2@W2 + b2 -> Q (WMMA)
    __syncthreads();
    ln_stats(Qb, muA, rsA, tid);
    __syncthreads();

    // ---- final LN (no activation) -> vectorized fp32 b128 stores to HBM ----
    for (int idx = tid; idx < ITEMS * J * (CH / 4); idx += NTHREADS) {
        const int item  = idx / (J * (CH / 4));
        const int r     = idx % (J * (CH / 4));
        const int joint = r / (CH / 4);
        const int c     = (r % (CH / 4)) * 4;
        const float mu = muA[item], rs = rsA[item];
        const v4bf   z = *(const v4bf*)&Qb[(item * RSTRIDE + joint) * CH + c];
        const float4 g = *(const float4*)&g2[joint * CH + c];
        const float4 b = *(const float4*)&be2[joint * CH + c];
        float4 o;
        o.x = ((float)z[0] - mu) * rs * g.x + b.x;
        o.y = ((float)z[1] - mu) * rs * g.y + b.y;
        o.z = ((float)z[2] - mu) * rs * g.z + b.z;
        o.w = ((float)z[3] - mu) * rs * g.w + b.w;
        *(float4*)&out[((size_t)(item0 + item) * J + joint) * CH + c] = o;
    }
}

extern "C" void kernel_launch(void* const* d_in, const int* in_sizes, int n_in,
                              void* d_out, int out_size, void* d_ws, size_t ws_size,
                              hipStream_t stream) {
    const float* skel = (const float*)d_in[0];
    const float* w_in = (const float*)d_in[1];
    const float* b_in = (const float*)d_in[2];
    const float* w1   = (const float*)d_in[3];
    const float* b1   = (const float*)d_in[4];
    const float* g1   = (const float*)d_in[5];
    const float* be1  = (const float*)d_in[6];
    const float* w2   = (const float*)d_in[7];
    const float* b2   = (const float*)d_in[8];
    const float* g2   = (const float*)d_in[9];
    const float* be2  = (const float*)d_in[10];
    float* out = (float*)d_out;

    __bf16* wsb = (__bf16*)d_ws;          // [0, MID*CH): W1 bf16; then CH*CH: W2 bf16
    cvt_weights<<<(CH * CH + 255) / 256, 256, 0, stream>>>(w1, w2, wsb);

    gcn_fused<<<B_TOTAL / ITEMS, NTHREADS, 0, stream>>>(
        skel, w_in, b_in,
        wsb, b1, g1, be1,
        wsb + MID * CH, b2, g2, be2,
        out);
}
